// RWKV_GPT_Block_20469814133539
// MI455X (gfx1250) — matrix operand order repaired and checked by reading, not compiled, not run on hardware
//
#include <hip/hip_runtime.h>
#include <hip/hip_bf16.h>
#include <stdint.h>

#define H_DIM 2048
#define H4_DIM 8192
#define T_LEN 2048
#define M_ROWS 8192   // B*T
#define EPS_LN 1e-5f

typedef __attribute__((ext_vector_type(16))) __bf16 v16bf;
typedef __attribute__((ext_vector_type(8)))  __bf16 v8bf;
typedef __attribute__((ext_vector_type(4)))  __bf16 v4bf;
typedef __attribute__((ext_vector_type(8)))  float  v8f;

__device__ __forceinline__ float dsigmoid(float v) { return 1.0f / (1.0f + __expf(-v)); }

// ---------------- fp32 -> bf16 weight conversion ----------------
__global__ __launch_bounds__(256)
void cvt_f32_bf16(const float* __restrict__ s, __bf16* __restrict__ d, int n)
{
    int i = (blockIdx.x * 256 + threadIdx.x) * 4;
    if (i + 3 < n) {
        float4 f = *(const float4*)(s + i);
        v4bf o;
        o.x = (__bf16)f.x; o.y = (__bf16)f.y; o.z = (__bf16)f.z; o.w = (__bf16)f.w;
        *(v4bf*)(d + i) = o;
    }
}

// ---------------- LayerNorm + time-shift + token mixes (fused) ----------------
// One block per (b,t) row. Computes LN of row t and row t-1, then up to three
// bf16 mixed outputs: out = h*mix + hh*(1-mix).  hh = 0 at t==0 (ZeroPad2d).
__global__ __launch_bounds__(256)
void ln_mix_kernel(const float* __restrict__ x,
                   const float* __restrict__ gamma, const float* __restrict__ beta,
                   const float* __restrict__ mk, const float* __restrict__ mv,
                   const float* __restrict__ mr,
                   __bf16* __restrict__ ok, __bf16* __restrict__ ov,
                   __bf16* __restrict__ orr)
{
    const int m   = blockIdx.x;
    const int t   = m & (T_LEN - 1);
    const int tid = threadIdx.x;
    const float* row  = x + (size_t)m * H_DIM;
    const bool hasPrev = (t != 0);
    const float* prev  = row - H_DIM;

    float v0[8], v1[8];
    float s0 = 0.f, q0 = 0.f, s1 = 0.f, q1 = 0.f;
#pragma unroll
    for (int i = 0; i < 8; ++i) {
        int c = tid + i * 256;
        float a = row[c];
        v0[i] = a; s0 += a; q0 += a * a;
        float b = hasPrev ? prev[c] : 0.f;
        v1[i] = b; s1 += b; q1 += b * b;
    }

    __shared__ float red[4][256];
    red[0][tid] = s0; red[1][tid] = q0; red[2][tid] = s1; red[3][tid] = q1;
    __syncthreads();
    for (int off = 128; off > 0; off >>= 1) {
        if (tid < off) {
            red[0][tid] += red[0][tid + off];
            red[1][tid] += red[1][tid + off];
            red[2][tid] += red[2][tid + off];
            red[3][tid] += red[3][tid + off];
        }
        __syncthreads();
    }
    const float inv = 1.0f / (float)H_DIM;
    float mean0 = red[0][0] * inv;
    float var0  = red[1][0] * inv - mean0 * mean0;
    float r0    = rsqrtf(var0 + EPS_LN);
    float mean1 = red[2][0] * inv;
    float var1  = red[3][0] * inv - mean1 * mean1;
    float r1    = rsqrtf(var1 + EPS_LN);

#pragma unroll
    for (int i = 0; i < 8; ++i) {
        int c = tid + i * 256;
        float g = gamma[c], b = beta[c];
        float h  = (v0[i] - mean0) * r0 * g + b;
        float hh = hasPrev ? ((v1[i] - mean1) * r1 * g + b) : 0.f;
        size_t idx = (size_t)m * H_DIM + c;
        float mkc = mk[c];
        ok[idx] = (__bf16)(h * mkc + hh * (1.f - mkc));
        if (ov) {
            float mvc = mv[c];
            ov[idx] = (__bf16)(h * mvc + hh * (1.f - mvc));
        }
        float mrc = mr[c];
        orr[idx] = (__bf16)(h * mrc + hh * (1.f - mrc));
    }
}

// ---------------- faithful wkv_run + sigmoid(r) gate -> bf16 ----------------
__global__ __launch_bounds__(256)
void wkv_gate(const float* __restrict__ k, const float* __restrict__ v,
              const float* __restrict__ r, const float* __restrict__ u,
              __bf16* __restrict__ g, int n)
{
    int i = (blockIdx.x * 256 + threadIdx.x) * 4;
    if (i + 3 < n) {
        float4 kk = *(const float4*)(k + i);
        float4 vv = *(const float4*)(v + i);
        float4 rr = *(const float4*)(r + i);
        int c = i & (H_DIM - 1);
        float4 uu = *(const float4*)(u + c);
        float ka[4] = {kk.x, kk.y, kk.z, kk.w};
        float va[4] = {vv.x, vv.y, vv.z, vv.w};
        float ra[4] = {rr.x, rr.y, rr.z, rr.w};
        float ua[4] = {uu.x, uu.y, uu.z, uu.w};
        __bf16 oa[4];
#pragma unroll
        for (int j = 0; j < 4; ++j) {
            // literal translation: aa=bb=0, pp=-1e38
            float ww = ua[j] + ka[j];
            float p  = fmaxf(-1e38f, ww);
            float e1 = __expf(-1e38f - p);       // -> 0
            float e2 = __expf(ww - p);           // -> 1
            float a  = e1 * 0.f + e2 * va[j];
            float bb = e1 * 0.f + e2;
            float wkv = a / bb;
            oa[j] = (__bf16)(dsigmoid(ra[j]) * wkv);
        }
        v4bf o = {oa[0], oa[1], oa[2], oa[3]};
        *(v4bf*)(g + i) = o;
    }
}

// ---------------- bf16 WMMA GEMM:  C[M,N] = A[M,K] . W[N,K]^T ----------------
// Double-buffered LDS staged with CDNA5 async global->LDS DMA (ASYNCcnt).
#define BM 128
#define BN 128
#define BKS 32
#define LDT 40  // padded LDS row stride (elements)

enum { EP_STORE_F32 = 0, EP_SILU_BF16 = 1, EP_ADD_F32 = 2, EP_FINAL = 3 };

__global__ __launch_bounds__(256)
void gemm_bf16_wmma(const __bf16* __restrict__ A, const __bf16* __restrict__ W,
                    int N, int K,
                    float* __restrict__ outF, __bf16* __restrict__ outB,
                    const float* __restrict__ aux1, const float* __restrict__ aux2,
                    int ep)
{
    __shared__ __bf16 sA[2][BM * LDT];
    __shared__ __bf16 sB[2][BN * LDT];

    const int tid   = threadIdx.x;
    const int lane  = tid & 31;
    const int wave  = tid >> 5;    // 0..7
    const int waveM = wave >> 2;   // 0..1  (64 rows of C)
    const int waveN = wave & 3;    // 0..3  (32 cols of C)
    const int bm = blockIdx.y * BM;
    const int bn = blockIdx.x * BN;
    const int lm = lane & 15;
    const int hs = lane >> 4;      // half-select per ISA WMMA lane layout

    // Per-thread cooperative-load coordinates: 2 x 16B chunks of each tile.
    const int c0 = tid,        r0 = c0 >> 2, kc0 = (c0 & 3) << 3;
    const int c1 = tid + 256,  r1 = c1 >> 2, kc1 = (c1 & 3) << 3;

    // Issue async DMA of one K-tile of A and W into LDS buffer `buf`.
    // VDST = per-lane LDS byte address (low 32 bits of shared-aperture addr),
    // VADDR = 64-bit global address, saddr = off (GV mode). Tracked by ASYNCcnt.
    auto issue_tile = [&](int buf, int kb) {
        unsigned lA0 = (unsigned)(uintptr_t)&sA[buf][r0 * LDT + kc0];
        unsigned lA1 = (unsigned)(uintptr_t)&sA[buf][r1 * LDT + kc1];
        unsigned lB0 = (unsigned)(uintptr_t)&sB[buf][r0 * LDT + kc0];
        unsigned lB1 = (unsigned)(uintptr_t)&sB[buf][r1 * LDT + kc1];
        const __bf16* gA0 = &A[(size_t)(bm + r0) * K + kb + kc0];
        const __bf16* gA1 = &A[(size_t)(bm + r1) * K + kb + kc1];
        const __bf16* gB0 = &W[(size_t)(bn + r0) * K + kb + kc0];
        const __bf16* gB1 = &W[(size_t)(bn + r1) * K + kb + kc1];
        asm volatile(
            "global_load_async_to_lds_b128 %0, %1, off\n\t"
            "global_load_async_to_lds_b128 %2, %3, off\n\t"
            "global_load_async_to_lds_b128 %4, %5, off\n\t"
            "global_load_async_to_lds_b128 %6, %7, off"
            :
            : "v"(lA0), "v"(gA0), "v"(lA1), "v"(gA1),
              "v"(lB0), "v"(gB0), "v"(lB1), "v"(gB1)
            : "memory");
    };

    v8f acc[4][2];
#pragma unroll
    for (int i = 0; i < 4; ++i)
#pragma unroll
        for (int j = 0; j < 2; ++j)
#pragma unroll
            for (int e = 0; e < 8; ++e) acc[i][j][e] = 0.0f;

    // Prologue: fill buffer 0.
    issue_tile(0, 0);
    asm volatile("s_wait_asynccnt 0x0" ::: "memory");
    __syncthreads();

    const int nk = K / BKS;
    for (int t = 0; t < nk; ++t) {
        const int cur = t & 1;
        if (t + 1 < nk) issue_tile(cur ^ 1, (t + 1) * BKS);

        // A fragment, 16x32 bf16. ISA layout: lane<16 (M=lane) holds K{0..7,16..23},
        // lane>=16 (M=lane-16) holds K{8..15,24..31}; 2 halves per VGPR.
        v16bf afrag[4];
#pragma unroll
        for (int i = 0; i < 4; ++i) {
            int row = waveM * 64 + i * 16 + lm;
            v8bf lo = *(const v8bf*)&sA[cur][row * LDT + (hs << 3)];
            v8bf hi = *(const v8bf*)&sA[cur][row * LDT + 16 + (hs << 3)];
            afrag[i] = __builtin_shufflevector(lo, hi,
                0,1,2,3,4,5,6,7,8,9,10,11,12,13,14,15);
        }
        // B fragment, 32x16 (KxN): lane<16 (N=lane) holds K{0..7,16..23}, lane>=16 K{8..15,24..31}.
        v16bf bfrag[2];
#pragma unroll
        for (int j = 0; j < 2; ++j) {
            int col = waveN * 32 + j * 16 + lm;
            v8bf lo = *(const v8bf*)&sB[cur][col * LDT + (hs << 3)];
            v8bf hi = *(const v8bf*)&sB[cur][col * LDT + 16 + (hs << 3)];
            bfrag[j] = __builtin_shufflevector(lo, hi,
                0,1,2,3,4,5,6,7,8,9,10,11,12,13,14,15);
        }

#pragma unroll
        for (int i = 0; i < 4; ++i)
#pragma unroll
            for (int j = 0; j < 2; ++j)
                acc[i][j] = __builtin_amdgcn_wmma_f32_16x16x32_bf16(
                    false, afrag[i], false, bfrag[j], (short)0, acc[i][j],
                    false, false);

        if (t + 1 < nk) {
            // Our DMA into the next buffer must be LDS-visible before any wave
            // reads it; the compiler's s_wait_dscnt-before-barrier covers the
            // WAR on the buffer we just read.
            asm volatile("s_wait_asynccnt 0x0" ::: "memory");
            __syncthreads();
        }
    }

    // Epilogue. D layout: VGPR e -> M = e + 8*(lane>=16), N = lane&15.
    const int mo = hs * 8;
#pragma unroll
    for (int i = 0; i < 4; ++i) {
#pragma unroll
        for (int j = 0; j < 2; ++j) {
            int col = bn + waveN * 32 + j * 16 + lm;
#pragma unroll
            for (int e = 0; e < 8; ++e) {
                int row = bm + waveM * 64 + i * 16 + mo + e;
                size_t idx = (size_t)row * N + col;
                float cval = acc[i][j][e];
                if (ep == EP_STORE_F32) {
                    outF[idx] = cval;
                } else if (ep == EP_SILU_BF16) {
                    outB[idx] = (__bf16)(cval * dsigmoid(cval));
                } else if (ep == EP_ADD_F32) {
                    outF[idx] = aux1[idx] + cval;
                } else { // EP_FINAL: x1 + sigmoid(cr) * kv
                    outF[idx] = aux1[idx] + dsigmoid(cval) * aux2[idx];
                }
            }
        }
    }
}

// ---------------- host-side orchestration ----------------
extern "C" void kernel_launch(void* const* d_in, const int* in_sizes, int n_in,
                              void* d_out, int out_size, void* d_ws, size_t ws_size,
                              hipStream_t stream)
{
    const float* x        = (const float*)d_in[0];
    const float* ln1_g    = (const float*)d_in[1];
    const float* ln1_b    = (const float*)d_in[2];
    const float* ln2_g    = (const float*)d_in[3];
    const float* ln2_b    = (const float*)d_in[4];
    /* d_in[5] tm_decay: unused by the literal wkv_run translation */
    const float* tm_first = (const float*)d_in[6];
    const float* tm_mk    = (const float*)d_in[7];
    const float* tm_mv    = (const float*)d_in[8];
    const float* tm_mr    = (const float*)d_in[9];
    const float* Wk       = (const float*)d_in[10];
    const float* Wv       = (const float*)d_in[11];
    const float* Wr       = (const float*)d_in[12];
    const float* Wo       = (const float*)d_in[13];
    const float* cm_mk    = (const float*)d_in[14];
    const float* cm_mr    = (const float*)d_in[15];
    const float* Wkey     = (const float*)d_in[16];
    const float* Wval     = (const float*)d_in[17];
    const float* Wcr      = (const float*)d_in[18];
    float* out = (float*)d_out;

    char* ws = (char*)d_ws;
    const size_t MB = 1024ull * 1024ull;
    __bf16* Wk_b   = (__bf16*)(ws +   0 * MB);   // 8 MB
    __bf16* Wv_b   = (__bf16*)(ws +   8 * MB);   // 8 MB
    __bf16* Wr_b   = (__bf16*)(ws +  16 * MB);   // 8 MB
    __bf16* Wo_b   = (__bf16*)(ws +  24 * MB);   // 8 MB
    __bf16* Wcr_b  = (__bf16*)(ws +  32 * MB);   // 8 MB
    __bf16* Wkey_b = (__bf16*)(ws +  40 * MB);   // 32 MB
    __bf16* Wval_b = (__bf16*)(ws +  72 * MB);   // 32 MB
    __bf16* actA   = (__bf16*)(ws + 104 * MB);   // 32 MB  ak -> g -> bk
    __bf16* actB   = (__bf16*)(ws + 136 * MB);   // 32 MB  av -> br
    __bf16* actC   = (__bf16*)(ws + 168 * MB);   // 32 MB  ar
    float*  f1     = (float*) (ws + 200 * MB);   // 64 MB  k  -> x1
    float*  f2     = (float*) (ws + 264 * MB);   // 64 MB  v  -> kv
    float*  f3     = (float*) (ws + 328 * MB);   // 64 MB  r
    __bf16* kks    = (__bf16*)(ws + 392 * MB);   // 128 MB silu(kk)

    const int HH  = H_DIM * H_DIM;   // 4M
    const int HH4 = H4_DIM * H_DIM;  // 16M
    cvt_f32_bf16<<<HH  / 1024, 256, 0, stream>>>(Wk,   Wk_b,   HH);
    cvt_f32_bf16<<<HH  / 1024, 256, 0, stream>>>(Wv,   Wv_b,   HH);
    cvt_f32_bf16<<<HH  / 1024, 256, 0, stream>>>(Wr,   Wr_b,   HH);
    cvt_f32_bf16<<<HH  / 1024, 256, 0, stream>>>(Wo,   Wo_b,   HH);
    cvt_f32_bf16<<<HH  / 1024, 256, 0, stream>>>(Wcr,  Wcr_b,  HH);
    cvt_f32_bf16<<<HH4 / 1024, 256, 0, stream>>>(Wkey, Wkey_b, HH4);
    cvt_f32_bf16<<<HH4 / 1024, 256, 0, stream>>>(Wval, Wval_b, HH4);

    // --- time-mix ---
    ln_mix_kernel<<<M_ROWS, 256, 0, stream>>>(x, ln1_g, ln1_b,
                                              tm_mk, tm_mv, tm_mr,
                                              actA, actB, actC);
    dim3 g1(H_DIM / BN, M_ROWS / BM);
    gemm_bf16_wmma<<<g1, 256, 0, stream>>>(actA, Wk_b, H_DIM, H_DIM,
                                           f1, nullptr, nullptr, nullptr, EP_STORE_F32);
    gemm_bf16_wmma<<<g1, 256, 0, stream>>>(actB, Wv_b, H_DIM, H_DIM,
                                           f2, nullptr, nullptr, nullptr, EP_STORE_F32);
    gemm_bf16_wmma<<<g1, 256, 0, stream>>>(actC, Wr_b, H_DIM, H_DIM,
                                           f3, nullptr, nullptr, nullptr, EP_STORE_F32);
    const int nMH = M_ROWS * H_DIM;  // 16M
    wkv_gate<<<nMH / 1024, 256, 0, stream>>>(f1, f2, f3, tm_first, actA, nMH);
    // x1 = x + g @ Wo^T   (into f1)
    gemm_bf16_wmma<<<g1, 256, 0, stream>>>(actA, Wo_b, H_DIM, H_DIM,
                                           f1, nullptr, x, nullptr, EP_ADD_F32);

    // --- channel-mix ---
    ln_mix_kernel<<<M_ROWS, 256, 0, stream>>>(f1, ln2_g, ln2_b,
                                              cm_mk, nullptr, cm_mr,
                                              actA, nullptr, actB);
    dim3 g2(H4_DIM / BN, M_ROWS / BM);
    // kks = silu(bk @ Wkey^T)  (bf16, M x 4H)
    gemm_bf16_wmma<<<g2, 256, 0, stream>>>(actA, Wkey_b, H4_DIM, H_DIM,
                                           nullptr, kks, nullptr, nullptr, EP_SILU_BF16);
    // kv = kks @ Wval^T        (f32, into f2)
    gemm_bf16_wmma<<<g1, 256, 0, stream>>>(kks, Wval_b, H_DIM, H4_DIM,
                                           f2, nullptr, nullptr, nullptr, EP_STORE_F32);
    // out = x1 + sigmoid(br @ Wcr^T) * kv
    gemm_bf16_wmma<<<g1, 256, 0, stream>>>(actB, Wcr_b, H_DIM, H_DIM,
                                           out, nullptr, f1, f2, EP_FINAL);
}
